// HDN_DDI_Block_68942815036028
// MI455X (gfx1250) — matrix operand order repaired and checked by reading, not compiled, not run on hardware
//
#include <hip/hip_runtime.h>

// ---------------- problem constants ----------------
#define NGRAPH 1024
#define PNODES 128
#define NNODES (NGRAPH * PNODES)   // 131072
#define NEDGE  1048576
#define DIN    128

typedef __attribute__((ext_vector_type(16))) __bf16 v16bf;
typedef __attribute__((ext_vector_type(8)))  float  v8f;

// ---------------- small device helpers ----------------
__device__ __forceinline__ float eluf(float x)   { return x > 0.f ? x : (__expf(x) - 1.f); }
__device__ __forceinline__ float lrelu(float x)  { return x > 0.f ? x : 0.2f * x; }

// monotone float->uint encoding so unsigned atomicMax == float max
__device__ __forceinline__ unsigned fenc(float f) {
  unsigned u = __float_as_uint(f);
  return (u & 0x80000000u) ? ~u : (u | 0x80000000u);
}
__device__ __forceinline__ float fdec(unsigned e) {
  unsigned u = (e & 0x80000000u) ? (e ^ 0x80000000u) : ~e;
  return __uint_as_float(u);
}
#define ENC_NEG_INF 0x007FFFFFu   // fenc(-inf)

// ---------------- elementwise / conversion kernels ----------------
__global__ void k_elu_cvt_bf16(const float* __restrict__ x, __bf16* __restrict__ y, long n) {
  long i = (long)blockIdx.x * blockDim.x + threadIdx.x;
  if (i < n) y[i] = (__bf16)eluf(x[i]);
}

__global__ void k_fill_u32(unsigned* __restrict__ p, unsigned v, int n) {
  int i = blockIdx.x * blockDim.x + threadIdx.x;
  if (i < n) p[i] = v;
}
__global__ void k_fill_f32(float* __restrict__ p, float v, int n) {
  int i = blockIdx.x * blockDim.x + threadIdx.x;
  if (i < n) p[i] = v;
}

// fold att vector into weight: V[k,h] = sum_c W[k, h*C + c] * att[h*C + c]
__global__ void k_fold_att(const float* __restrict__ W, const float* __restrict__ att,
                           float* __restrict__ V, int C) {
  int i = blockIdx.x * blockDim.x + threadIdx.x; // k*2 + h
  if (i >= DIN * 2) return;
  int k = i >> 1, h = i & 1;
  float s = 0.f;
  for (int c = 0; c < C; ++c) s += W[k * (2 * C) + h * C + c] * att[h * C + c];
  V[k * 2 + h] = s;
}

// per-node attention logits: A[n,h] = sum_k f(x[n,k]) * V[k,h]   (f = elu or id)
__global__ void k_a_compute(const float* __restrict__ X, const float* __restrict__ V,
                            float* __restrict__ A, int applyElu) {
  int n = blockIdx.x * blockDim.x + threadIdx.x;
  if (n >= NNODES) return;
  const float* x = X + (size_t)n * DIN;
  float a0 = 0.f, a1 = 0.f;
  for (int k = 0; k < DIN; ++k) {
    float v = x[k];
    if (applyElu) v = eluf(v);
    a0 += v * V[k * 2 + 0];
    a1 += v * V[k * 2 + 1];
  }
  A[n * 2 + 0] = a0;
  A[n * 2 + 1] = a1;
}

// ---------------- B-fragment pre-pack -------------------------------------------
// Swizzle W[128,NC] (f32, row-major) into per-(kblock, coltile, lane) contiguous
// bf16 fragments matching the ISA 16-bit B 32x16 lane layout, so the GEMM can do
// one aligned 32-byte vector load per lane per K-step.
//   Wp[(((kb*tilesN)+tn)*32 + lane)*16 + i] = W[(kb*32 + (lane<16?0:8) + (i&7) + (i&8?16:0))*NC
//                                               + tn*16 + (lane&15)]
__global__ void k_pack_w(const float* __restrict__ W, __bf16* __restrict__ Wp, int NC) {
  const int tilesN = NC >> 4;
  const int total  = 4 * tilesN * 32 * 16;   // kblocks * tiles * lanes * elems
  int idx = blockIdx.x * blockDim.x + threadIdx.x;
  if (idx >= total) return;
  int i  = idx & 15;
  int l  = (idx >> 4) & 31;
  int tn = (idx >> 9) % tilesN;
  int kb = idx / (512 * tilesN);
  int col = tn * 16 + (l & 15);
  int k   = kb * 32 + ((l < 16) ? 0 : 8) + (i & 7) + ((i & 8) ? 16 : 0);
  Wp[idx] = (__bf16)W[(size_t)k * NC + col];
}

// ---------------- WMMA GEMM: Y[N,NC] = Xbf[N,128] @ W[128,NC], f32 accum -----------
// one 16x16 C tile per wave32; K=128 as 4 steps of v_wmma_f32_16x16x32_bf16.
// A: two b128 loads per step straight from the row; B: one aligned v16bf load
// from the pre-packed fragment table.
__global__ void k_gemm_bf16_wmma(const __bf16* __restrict__ X, const __bf16* __restrict__ Wp,
                                 float* __restrict__ Y, int NC) {
  const int lane = threadIdx.x;          // 0..31
  const int tilesN = NC >> 4;
  const int tile = blockIdx.x * blockDim.y + threadIdx.y;
  const int nTiles = (NNODES >> 4) * tilesN;
  if (tile >= nTiles) return;            // wave-uniform
  const int tm = tile / tilesN, tn = tile % tilesN;

  const int row   = tm * 16 + (lane & 15);
  const int abase = (lane < 16) ? 0 : 8;     // ISA 16-bit A 16x32 layout

  v8f acc = {};
#pragma unroll
  for (int kb = 0; kb < 4; ++kb) {
    const int k0 = kb * 32;
    v16bf a;
    const __bf16* xp = X + (size_t)row * DIN + k0 + abase;
#pragma unroll
    for (int i = 0; i < 8; ++i) { a[i] = xp[i]; a[i + 8] = xp[16 + i]; }
    const v16bf b = *(const v16bf*)(Wp + ((((size_t)kb * tilesN) + tn) * 32 + lane) * 16);
    acc = __builtin_amdgcn_wmma_f32_16x16x32_bf16(false, a, false, b,
                                                  (short)0, acc, false, false);
  }
  const int srow = tm * 16 + ((lane < 16) ? 0 : 8);
  const int col  = tn * 16 + (lane & 15);
#pragma unroll
  for (int r = 0; r < 8; ++r) Y[(size_t)(srow + r) * NC + col] = acc[r];
}

// ---------------- segment softmax passes ----------------
__global__ void k_segmax_edges(const int* __restrict__ src, const int* __restrict__ dst,
                               const float* __restrict__ As, const float* __restrict__ Ad,
                               unsigned* __restrict__ Menc, int nE) {
  int e = blockIdx.x * blockDim.x + threadIdx.x;
  if (e >= nE) return;
  int s = src[e], d = dst[e];
#pragma unroll
  for (int h = 0; h < 2; ++h) {
    float l = lrelu(As[s * 2 + h] + Ad[d * 2 + h]);
    atomicMax(&Menc[d * 2 + h], fenc(l));
  }
}
__global__ void k_segmax_loops(const float* __restrict__ As, const float* __restrict__ Ad,
                               unsigned* __restrict__ Menc) {
  int n = blockIdx.x * blockDim.x + threadIdx.x;
  if (n >= NNODES) return;
#pragma unroll
  for (int h = 0; h < 2; ++h) {
    float l = lrelu(As[n * 2 + h] + Ad[n * 2 + h]);
    atomicMax(&Menc[n * 2 + h], fenc(l));
  }
}

__global__ void k_decode_m(const unsigned* __restrict__ Menc, float* __restrict__ Mf, int n) {
  int i = blockIdx.x * blockDim.x + threadIdx.x;
  if (i >= n) return;
  float f = fdec(Menc[i]);
  unsigned u = __float_as_uint(f);
  // m = where(isfinite(m), m, 0)
  Mf[i] = ((u & 0x7f800000u) == 0x7f800000u) ? 0.f : f;
}

__global__ void k_segsum_edges(const int* __restrict__ src, const int* __restrict__ dst,
                               const float* __restrict__ As, const float* __restrict__ Ad,
                               const float* __restrict__ Mf, float* __restrict__ Ssum, int nE) {
  int e = blockIdx.x * blockDim.x + threadIdx.x;
  if (e >= nE) return;
  int s = src[e], d = dst[e];
#pragma unroll
  for (int h = 0; h < 2; ++h) {
    float l = lrelu(As[s * 2 + h] + Ad[d * 2 + h]);
    atomicAdd(&Ssum[d * 2 + h], __expf(l - Mf[d * 2 + h]));
  }
}
__global__ void k_segsum_loops(const float* __restrict__ As, const float* __restrict__ Ad,
                               const float* __restrict__ Mf, float* __restrict__ Ssum) {
  int n = blockIdx.x * blockDim.x + threadIdx.x;
  if (n >= NNODES) return;
#pragma unroll
  for (int h = 0; h < 2; ++h) {
    float l = lrelu(As[n * 2 + h] + Ad[n * 2 + h]);
    atomicAdd(&Ssum[n * 2 + h], __expf(l - Mf[n * 2 + h]));
  }
}

__global__ void k_init_bias(float* __restrict__ Out, int ldo, int colOff,
                            const float* __restrict__ bias, int NC) {
  long i = (long)blockIdx.x * blockDim.x + threadIdx.x;
  if (i >= (long)NNODES * NC) return;
  int n = (int)(i / NC), j = (int)(i % NC);
  Out[(size_t)n * ldo + colOff + j] = bias[j];
}

// channel-parallel weighted scatter: Out[dst, colOff+j] += alpha(e,h) * HS[src, j]
__global__ void k_scatter_edges(const int* __restrict__ src, const int* __restrict__ dst,
                                const float* __restrict__ As, const float* __restrict__ Ad,
                                const float* __restrict__ Mf, const float* __restrict__ Ssum,
                                const float* __restrict__ HS, float* __restrict__ Out,
                                int ldo, int colOff, int C, int nE) {
  const int HC  = 2 * C;
  const int epb = blockDim.x / HC;
  const int e = blockIdx.x * epb + (int)(threadIdx.x / HC);
  const int j = threadIdx.x % HC;
  if (e >= nE) return;
  int s = src[e], d = dst[e];
  int h = j / C;
  float l = lrelu(As[s * 2 + h] + Ad[d * 2 + h]);
  float alpha = __expf(l - Mf[d * 2 + h]) / (Ssum[d * 2 + h] + 1e-16f);
  atomicAdd(&Out[(size_t)d * ldo + colOff + j], alpha * HS[(size_t)s * HC + j]);
}
__global__ void k_scatter_loops(const float* __restrict__ As, const float* __restrict__ Ad,
                                const float* __restrict__ Mf, const float* __restrict__ Ssum,
                                const float* __restrict__ HS, float* __restrict__ Out,
                                int ldo, int colOff, int C) {
  const int HC  = 2 * C;
  const int npb = blockDim.x / HC;
  const int n = blockIdx.x * npb + (int)(threadIdx.x / HC);
  const int j = threadIdx.x % HC;
  if (n >= NNODES) return;
  int h = j / C;
  float l = lrelu(As[n * 2 + h] + Ad[n * 2 + h]);
  float alpha = __expf(l - Mf[n * 2 + h]) / (Ssum[n * 2 + h] + 1e-16f);
  atomicAdd(&Out[(size_t)n * ldo + colOff + j], alpha * HS[(size_t)n * HC + j]);
}

// ---------------- per-graph LayerNorm + ELU (in place), emit bf16 copy --------------
__global__ void k_ln_graph_elu(float* __restrict__ X, __bf16* __restrict__ Xb,
                               const float* __restrict__ lw, const float* __restrict__ lb) {
  __shared__ float sS[256], sQ[256];
  const int g = blockIdx.x;
  float* xg = X + (size_t)g * PNODES * DIN;
  float s = 0.f, q = 0.f;
  for (int i = threadIdx.x; i < PNODES * DIN; i += blockDim.x) {
    float v = xg[i]; s += v; q += v * v;
  }
  sS[threadIdx.x] = s; sQ[threadIdx.x] = q;
  __syncthreads();
  for (int st = 128; st > 0; st >>= 1) {
    if ((int)threadIdx.x < st) { sS[threadIdx.x] += sS[threadIdx.x + st];
                                 sQ[threadIdx.x] += sQ[threadIdx.x + st]; }
    __syncthreads();
  }
  const float inv_n = 1.f / (PNODES * DIN);
  float mean = sS[0] * inv_n;
  float var  = sQ[0] * inv_n - mean * mean;
  float inv  = rsqrtf(var + 1e-5f);
  for (int i = threadIdx.x; i < PNODES * DIN; i += blockDim.x) {
    int c = i & (DIN - 1);
    float v = (xg[i] - mean) * inv * lw[c] + lb[c];
    v = eluf(v);
    xg[i] = v;
    Xb[(size_t)g * PNODES * DIN + i] = (__bf16)v;
  }
}

// gather per-graph supernode (y==2; exactly one per graph, graphs contiguous)
__global__ void k_select(const float* __restrict__ Out, const int* __restrict__ y,
                         float* __restrict__ Sel) {
  int n = blockIdx.x * blockDim.x + threadIdx.x;
  if (n >= NNODES) return;
  if (y[n] == 2) {
    int g = n / PNODES;
    for (int c = 0; c < DIN; ++c) Sel[(size_t)g * DIN + c] = Out[(size_t)n * DIN + c];
  }
}

// ---------------- host-side GAT driver ----------------
static void run_gat(hipStream_t st,
                    const float* XsrcF, int eluS, const float* XdstF, int eluD,
                    const __bf16* XsrcB, const __bf16* Wpacked,
                    const float* Vsrc, const float* Vdst, const float* bias,
                    int C, const int* src, const int* dst, int nE, bool loops,
                    float* Out, int ldo, int colOff,
                    float* AS, float* AD, unsigned* MENC, float* MF, float* SSUM, float* HS) {
  const int NC = 2 * C;
  k_a_compute<<<NNODES / 256, 256, 0, st>>>(XsrcF, Vsrc, AS, eluS);
  k_a_compute<<<NNODES / 256, 256, 0, st>>>(XdstF, Vdst, AD, eluD);
  {
    int nTiles = (NNODES / 16) * (NC / 16);
    k_gemm_bf16_wmma<<<nTiles / 4, dim3(32, 4), 0, st>>>(XsrcB, Wpacked, HS, NC);
  }
  k_fill_u32<<<(NNODES * 2) / 256, 256, 0, st>>>(MENC, ENC_NEG_INF, NNODES * 2);
  k_fill_f32<<<(NNODES * 2) / 256, 256, 0, st>>>(SSUM, 0.f, NNODES * 2);
  k_segmax_edges<<<nE / 256, 256, 0, st>>>(src, dst, AS, AD, MENC, nE);
  if (loops) k_segmax_loops<<<NNODES / 256, 256, 0, st>>>(AS, AD, MENC);
  k_decode_m<<<(NNODES * 2) / 256, 256, 0, st>>>(MENC, MF, NNODES * 2);
  k_segsum_edges<<<nE / 256, 256, 0, st>>>(src, dst, AS, AD, MF, SSUM, nE);
  if (loops) k_segsum_loops<<<NNODES / 256, 256, 0, st>>>(AS, AD, MF, SSUM);
  k_init_bias<<<(int)(((long)NNODES * NC) / 256), 256, 0, st>>>(Out, ldo, colOff, bias, NC);
  {
    int HC = 2 * C;
    k_scatter_edges<<<(int)(((long)nE * HC) / 256), 256, 0, st>>>(
        src, dst, AS, AD, MF, SSUM, HS, Out, ldo, colOff, C, nE);
    if (loops)
      k_scatter_loops<<<(int)(((long)NNODES * HC) / 256), 256, 0, st>>>(
          AS, AD, MF, SSUM, HS, Out, ldo, colOff, C);
  }
}

extern "C" void kernel_launch(void* const* d_in, const int* in_sizes, int n_in,
                              void* d_out, int out_size, void* d_ws, size_t ws_size,
                              hipStream_t stream) {
  // -------- inputs (setup_inputs order) --------
  const float* h_x = (const float*)d_in[0];
  const float* t_x = (const float*)d_in[1];
  const int* h_ei  = (const int*)d_in[2];
  const int* t_ei  = (const int*)d_in[3];
  const int* b_ei  = (const int*)d_in[4];
  const int* h_y   = (const int*)d_in[7];
  const int* t_y   = (const int*)d_in[8];
  const float* w_intra   = (const float*)d_in[9];
  const float* att_in_s  = (const float*)d_in[10];
  const float* att_in_d  = (const float*)d_in[11];
  const float* b_intra   = (const float*)d_in[12];
  const float* w_int_s   = (const float*)d_in[13];
  const float* w_int_d   = (const float*)d_in[14];
  const float* att_it_s  = (const float*)d_in[15];
  const float* att_it_d  = (const float*)d_in[16];
  const float* b_inter   = (const float*)d_in[17];
  const float* w_pool    = (const float*)d_in[18];
  const float* att_pl_s  = (const float*)d_in[19];
  const float* att_pl_d  = (const float*)d_in[20];
  const float* b_pool    = (const float*)d_in[21];
  const float* ln_w      = (const float*)d_in[22];
  const float* ln_b      = (const float*)d_in[23];

  // -------- outputs --------
  float* HOUT = (float*)d_out;
  float* TOUT = HOUT + (size_t)NNODES * DIN;
  float* HSEL = TOUT + (size_t)NNODES * DIN;
  float* TSEL = HSEL + (size_t)NGRAPH * DIN;

  // -------- workspace bump allocator --------
  char* ws = (char*)d_ws;
  size_t off = 0;
  auto alloc = [&](size_t bytes) -> void* {
    void* p = ws + off;
    off += (bytes + 255) & ~(size_t)255;
    return p;
  };
  __bf16* HXB   = (__bf16*)alloc((size_t)NNODES * DIN * 2);   // elu(h_x) bf16, later h_nx bf16
  __bf16* TXB   = (__bf16*)alloc((size_t)NNODES * DIN * 2);
  float*  HREP  = (float*) alloc((size_t)NNODES * DIN * 4);   // [intra | inter] -> h_nx in place
  float*  TREP  = (float*) alloc((size_t)NNODES * DIN * 4);
  float*  HS    = (float*) alloc((size_t)NNODES * DIN * 4);   // GEMM output (<= N x 128)
  float*  AS    = (float*) alloc((size_t)NNODES * 2 * 4);
  float*  AD    = (float*) alloc((size_t)NNODES * 2 * 4);
  unsigned* MENC= (unsigned*)alloc((size_t)NNODES * 2 * 4);
  float*  MF    = (float*) alloc((size_t)NNODES * 2 * 4);
  float*  SSUM  = (float*) alloc((size_t)NNODES * 2 * 4);
  __bf16* WAP   = (__bf16*)alloc((size_t)DIN * 64 * 2);       // w_intra packed frags (NC=64: 8192)
  __bf16* WSP   = (__bf16*)alloc((size_t)DIN * 64 * 2);       // w_inter_src packed
  __bf16* WPP   = (__bf16*)alloc((size_t)DIN * 128 * 2);      // w_pool packed (NC=128: 16384)
  float*  V_in_s = (float*)alloc(DIN * 2 * 4);
  float*  V_in_d = (float*)alloc(DIN * 2 * 4);
  float*  V_it_s = (float*)alloc(DIN * 2 * 4);
  float*  V_it_d = (float*)alloc(DIN * 2 * 4);
  float*  V_pl_s = (float*)alloc(DIN * 2 * 4);
  float*  V_pl_d = (float*)alloc(DIN * 2 * 4);
  (void)ws_size; (void)n_in; (void)in_sizes; (void)out_size;

  const int* h_src = h_ei;        const int* h_dst = h_ei + NEDGE;
  const int* t_src = t_ei;        const int* t_dst = t_ei + NEDGE;
  const int* b_src = b_ei;        const int* b_dst = b_ei + NEDGE;

  // -------- phase 0: conversions + packed B fragments + folded attention vectors ----
  k_elu_cvt_bf16<<<(NNODES * DIN) / 256, 256, 0, stream>>>(h_x, HXB, (long)NNODES * DIN);
  k_elu_cvt_bf16<<<(NNODES * DIN) / 256, 256, 0, stream>>>(t_x, TXB, (long)NNODES * DIN);
  k_pack_w<<<(64 * 128) / 256, 256, 0, stream>>>(w_intra, WAP, 64);
  k_pack_w<<<(64 * 128) / 256, 256, 0, stream>>>(w_int_s, WSP, 64);
  k_pack_w<<<(128 * 128) / 256, 256, 0, stream>>>(w_pool, WPP, 128);
  k_fold_att<<<1, 256, 0, stream>>>(w_intra, att_in_s, V_in_s, 32);
  k_fold_att<<<1, 256, 0, stream>>>(w_intra, att_in_d, V_in_d, 32);
  k_fold_att<<<1, 256, 0, stream>>>(w_int_s, att_it_s, V_it_s, 32);
  k_fold_att<<<1, 256, 0, stream>>>(w_int_d, att_it_d, V_it_d, 32);
  k_fold_att<<<1, 256, 0, stream>>>(w_pool, att_pl_s, V_pl_s, 64);
  k_fold_att<<<1, 256, 0, stream>>>(w_pool, att_pl_d, V_pl_d, 64);

  // -------- intra GATs (self loops) -> rep[:, 0:64] --------
  run_gat(stream, h_x, 1, h_x, 1, HXB, WAP, V_in_s, V_in_d, b_intra, 32,
          h_src, h_dst, NEDGE, true, HREP, DIN, 0, AS, AD, MENC, MF, SSUM, HS);
  run_gat(stream, t_x, 1, t_x, 1, TXB, WAP, V_in_s, V_in_d, b_intra, 32,
          t_src, t_dst, NEDGE, true, TREP, DIN, 0, AS, AD, MENC, MF, SSUM, HS);

  // -------- inter (bipartite, no self loops) -> rep[:, 64:128] --------
  // t_inter: src = h graph, dst = t graph, edges as-is
  run_gat(stream, h_x, 1, t_x, 1, HXB, WSP, V_it_s, V_it_d, b_inter, 32,
          b_src, b_dst, NEDGE, false, TREP, DIN, 64, AS, AD, MENC, MF, SSUM, HS);
  // h_inter: src = t graph, dst = h graph, edges reversed
  run_gat(stream, t_x, 1, h_x, 1, TXB, WSP, V_it_s, V_it_d, b_inter, 32,
          b_dst, b_src, NEDGE, false, HREP, DIN, 64, AS, AD, MENC, MF, SSUM, HS);

  // -------- per-graph LayerNorm + ELU (in place), fresh bf16 copies --------
  k_ln_graph_elu<<<NGRAPH, 256, 0, stream>>>(HREP, HXB, ln_w, ln_b);
  k_ln_graph_elu<<<NGRAPH, 256, 0, stream>>>(TREP, TXB, ln_w, ln_b);

  // -------- pool GATs (self loops) -> d_out --------
  run_gat(stream, HREP, 0, HREP, 0, HXB, WPP, V_pl_s, V_pl_d, b_pool, 64,
          h_src, h_dst, NEDGE, true, HOUT, DIN, 0, AS, AD, MENC, MF, SSUM, HS);
  run_gat(stream, TREP, 0, TREP, 0, TXB, WPP, V_pl_s, V_pl_d, b_pool, 64,
          t_src, t_dst, NEDGE, true, TOUT, DIN, 0, AS, AD, MENC, MF, SSUM, HS);

  // -------- supernode gathers --------
  k_select<<<NNODES / 256, 256, 0, stream>>>(HOUT, h_y, HSEL);
  k_select<<<NNODES / 256, 256, 0, stream>>>(TOUT, t_y, TSEL);
}
